// TemporalMultiheadAttention_16844861734929
// MI455X (gfx1250) — matrix operand order repaired and checked by reading, not compiled, not run on hardware
//
#include <hip/hip_runtime.h>
#include <hip/hip_bf16.h>

// ---------------------------------------------------------------------------
// Temporal MHA for MI455X (gfx1250, wave32, WMMA 16x16x32 bf16 / f32 acc)
// L=1024, N=8, E=512, H=8, hd=64.  R = L*N = 8192 rows.
// alpha_time factorizes rank-1: folded into Q/K row scaling at projection.
// WMMA loops are software-pipelined; sched_group_barrier directives pin the
// load/WMMA interleave so the backend scheduler cannot collapse the pipeline
// back into load->wait(0)->wmma chains.
// ---------------------------------------------------------------------------

#define LL 1024
#define NN 8
#define EE 512
#define HH 8
#define HD 64
#define RR (LL * NN)        // 8192
#define E3 (3 * EE)         // 1536

typedef __bf16 bf16_t;
typedef __attribute__((ext_vector_type(16))) __bf16 v16bf;
typedef __attribute__((ext_vector_type(8)))  __bf16 v8bf;
typedef __attribute__((ext_vector_type(8)))  float  v8f;

// ---- scheduling directives -------------------------------------------------
// masks: 0x008 = MFMA/WMMA, 0x020 = VMEM read, 0x100 = DS read, 0x200 = DS write
#if defined(__has_builtin)
#  if __has_builtin(__builtin_amdgcn_sched_group_barrier)
#    define SGB(mask, cnt) __builtin_amdgcn_sched_group_barrier((mask), (cnt), 0)
#  endif
#  if __has_builtin(__builtin_amdgcn_sched_barrier)
#    define SFENCE() __builtin_amdgcn_sched_barrier(0)
#  endif
#endif
#ifndef SGB
#  define SGB(mask, cnt)
#endif
#ifndef SFENCE
#  define SFENCE()
#endif

// ---- WMMA wrapper ----------------------------------------------------------
__device__ __forceinline__ v8f wmma_bf16(v16bf a, v16bf b, v8f c) {
  return __builtin_amdgcn_wmma_f32_16x16x32_bf16(
      /*neg_a=*/false, a, /*neg_b=*/false, b,
      /*c_mod=*/(short)0, c, /*reuse_a=*/false, /*reuse_b=*/false);
}

// ---- Fragment loads (16x16x32 bf16 layouts, wave32) ------------------------
// A (16xK row-major, stride lda): lane m = lane&15, hf = lane>>4.
//   v0..3 hold K = k0 + hf*8 + [0..7] ; v4..7 hold K = k0 + 16 + hf*8 + [0..7]
__device__ __forceinline__ v16bf load_frag_a(const bf16_t* A, int lda,
                                             int m, int hf, int k0) {
  const bf16_t* p0 = A + m * lda + k0 + hf * 8;
  v8bf lo = *(const v8bf*)(p0);
  v8bf hi = *(const v8bf*)(p0 + 16);
  v16bf a;
#pragma unroll
  for (int i = 0; i < 8; ++i) { a[i] = lo[i]; a[8 + i] = hi[i]; }
  return a;
}

// B (KxN), stored transposed as Bt[n][k] row-major with stride ldb:
//   lane n = lane&15, hf = lane>>4: v0..3 hold K = k0 + hf*8 + [0..7] ; v4..7 hold K = k0 + 16 + hf*8 + [0..7]
__device__ __forceinline__ v16bf load_frag_b(const bf16_t* Bt, int ldb,
                                             int n, int hf, int k0) {
  const bf16_t* p = Bt + n * ldb + k0 + hf * 8;
  v8bf lo = *(const v8bf*)(p);
  v8bf hi = *(const v8bf*)(p + 16);
  v16bf b;
#pragma unroll
  for (int i = 0; i < 8; ++i) { b[i] = lo[i]; b[8 + i] = hi[i]; }
  return b;
}

// ---------------------------------------------------------------------------
// Kernel 1: convert f32 -> bf16 (query, W_in, W_out) + temporal scale factors
// ---------------------------------------------------------------------------
#define N_XB   (RR * EE)       // 4194304
#define N_WIN  (E3 * EE)       // 786432
#define N_WOUT (EE * EE)       // 262144
#define N_ET   (RR)            // 8192
#define N_CVT  (N_XB + N_WIN + N_WOUT + N_ET)

__global__ __launch_bounds__(256) void convert_kernel(
    const float* __restrict__ query, const float* __restrict__ W_in,
    const float* __restrict__ W_out, const float* __restrict__ timev,
    bf16_t* __restrict__ XB, bf16_t* __restrict__ WinB,
    bf16_t* __restrict__ WoutB, float* __restrict__ ep, float* __restrict__ en) {
  int idx = blockIdx.x * 256 + threadIdx.x;
  if (idx < N_XB) {
    XB[idx] = (bf16_t)query[idx];
  } else if (idx < N_XB + N_WIN) {
    int i = idx - N_XB;
    WinB[i] = (bf16_t)W_in[i];
  } else if (idx < N_XB + N_WIN + N_WOUT) {
    int i = idx - N_XB - N_WIN;
    WoutB[i] = (bf16_t)W_out[i];
  } else if (idx < N_CVT) {
    int i = idx - N_XB - N_WIN - N_WOUT;        // i = l*N + b
    float t = timev[i] * 1.0e-6f;
    ep[i] = __expf(t);                           // folded into Q rows
    en[i] = __expf(-t);                          // folded into K rows
  }
}

// ---------------------------------------------------------------------------
// Kernel 2: QKV projection. Each wave owns a 16x64 strip of (8192 x 1536).
// Software-pipelined over K with pinned load/WMMA interleave. Epilogue
// scatters into per-(b,h) Q (Lx64), K (Lx64), Vt (64xL), bf16.
// ---------------------------------------------------------------------------
__global__ __launch_bounds__(256) void qkv_gemm_kernel(
    const bf16_t* __restrict__ XB, const bf16_t* __restrict__ WinB,
    const float* __restrict__ bias, const float* __restrict__ ep,
    const float* __restrict__ en, bf16_t* __restrict__ Q,
    bf16_t* __restrict__ K, bf16_t* __restrict__ Vt) {
  int wave = blockIdx.x * 8 + (threadIdx.x >> 5);
  int lane = threadIdx.x & 31;
  int nl = lane & 15, hf = lane >> 4;
  int mtile = wave / 24;              // 512 m-tiles
  int strip = wave % 24;              // 24 strips of 64 cols
  int m0 = mtile * 16, c0 = strip * 64;

  const bf16_t* Ab = XB + (size_t)m0 * EE;
  const bf16_t* Bb0 = WinB + (size_t)(c0 +  0) * EE;
  const bf16_t* Bb1 = WinB + (size_t)(c0 + 16) * EE;
  const bf16_t* Bb2 = WinB + (size_t)(c0 + 32) * EE;
  const bf16_t* Bb3 = WinB + (size_t)(c0 + 48) * EE;

  v8f acc0 = {}, acc1 = {}, acc2 = {}, acc3 = {};

  // prologue: fragments for k0 = 0
  v16bf a  = load_frag_a(Ab, EE, nl, hf, 0);
  v16bf b0 = load_frag_b(Bb0, EE, nl, hf, 0);
  v16bf b1 = load_frag_b(Bb1, EE, nl, hf, 0);
  v16bf b2 = load_frag_b(Bb2, EE, nl, hf, 0);
  v16bf b3 = load_frag_b(Bb3, EE, nl, hf, 0);
  SFENCE();

#pragma unroll
  for (int k0 = 32; k0 < EE; k0 += 32) {
    // prefetch next k-step into fresh registers (hides load latency)
    v16bf an  = load_frag_a(Ab, EE, nl, hf, k0);
    v16bf bn0 = load_frag_b(Bb0, EE, nl, hf, k0);
    v16bf bn1 = load_frag_b(Bb1, EE, nl, hf, k0);
    v16bf bn2 = load_frag_b(Bb2, EE, nl, hf, k0);
    v16bf bn3 = load_frag_b(Bb3, EE, nl, hf, k0);
    acc0 = wmma_bf16(a, b0, acc0);
    acc1 = wmma_bf16(a, b1, acc1);
    acc2 = wmma_bf16(a, b2, acc2);
    acc3 = wmma_bf16(a, b3, acc3);
    a = an; b0 = bn0; b1 = bn1; b2 = bn2; b3 = bn3;
    // pin interleave: spread the 10 prefetch loads between the 4 WMMAs
    SGB(0x020, 2); SGB(0x008, 1);
    SGB(0x020, 2); SGB(0x008, 1);
    SGB(0x020, 2); SGB(0x008, 1);
    SGB(0x020, 2); SGB(0x008, 1);
    SGB(0x020, 2);
  }
  acc0 = wmma_bf16(a, b0, acc0);
  acc1 = wmma_bf16(a, b1, acc1);
  acc2 = wmma_bf16(a, b2, acc2);
  acc3 = wmma_bf16(a, b3, acc3);

  v8f accs[4] = {acc0, acc1, acc2, acc3};
#pragma unroll
  for (int t = 0; t < 4; ++t) {
    int c = c0 + t * 16 + nl;
    int chunk = c / EE;               // 0=q 1=k 2=v
    int e = c % EE;
    int h = e >> 6, d = e & 63;
    float bv = bias[c];
#pragma unroll
    for (int r = 0; r < 8; ++r) {
      int row = m0 + hf * 8 + r;      // row = l*N + b
      int l = row >> 3, b_ = row & 7;
      int bh = b_ * HH + h;
      float v = accs[t][r] + bv;
      if (chunk == 0) {
        v *= 0.125f * ep[row];        // scaling * exp(+t/1e6)
        Q[((size_t)bh * LL + l) * HD + d] = (bf16_t)v;
      } else if (chunk == 1) {
        v *= en[row];                 // exp(-t/1e6)
        K[((size_t)bh * LL + l) * HD + d] = (bf16_t)v;
      } else {
        Vt[((size_t)bh * HD + d) * LL + l] = (bf16_t)v;  // transposed V
      }
    }
  }
}

// ---------------------------------------------------------------------------
// Kernel 3: attention. 1 block (128 thr = 4 waves) per (b,h, 16-row q-tile).
// LDS: 16x1024 f32 scores (64KB) + 16x1024 bf16 probs (32KB) + 128 f32 red.
// ---------------------------------------------------------------------------
#define SMEM_SC   (16 * 1024 * 4)
#define SMEM_P    (16 * 1024 * 2)
#define SMEM_RED  (128 * 4)
#define SMEM_ATTN (SMEM_SC + SMEM_P + SMEM_RED)

__global__ __launch_bounds__(128) void attn_kernel(
    const bf16_t* __restrict__ Q, const bf16_t* __restrict__ K,
    const bf16_t* __restrict__ Vt, bf16_t* __restrict__ AO) {
  extern __shared__ char smem[];
  float*  sc  = (float*)smem;                         // [16][1024]
  bf16_t* P   = (bf16_t*)(smem + SMEM_SC);            // [16][1024]
  float*  red = (float*)(smem + SMEM_SC + SMEM_P);    // [16][8]

  int bh = blockIdx.x >> 6;           // 64 (b,h) pairs
  int qt = blockIdx.x & 63;           // 64 q-tiles
  int b_ = bh >> 3, h = bh & 7;
  int wv = threadIdx.x >> 5, lane = threadIdx.x & 31;
  int nl = lane & 15, hf = lane >> 4;

  const bf16_t* Qb = Q  + (size_t)bh * LL * HD + (size_t)qt * 16 * HD;
  const bf16_t* Kb = K  + (size_t)bh * LL * HD;
  const bf16_t* Vb = Vt + (size_t)bh * HD * LL;

  // ---- Stage 1: scores = Q Kt (Q pre-scaled; alpha folded in) ----
  v16bf aq0 = load_frag_a(Qb, HD, nl, hf, 0);
  v16bf aq1 = load_frag_a(Qb, HD, nl, hf, 32);
  // pipelined over the 16 s-tiles this wave owns
  v16bf bk0 = load_frag_b(Kb + (size_t)wv * 16 * HD, HD, nl, hf, 0);
  v16bf bk1 = load_frag_b(Kb + (size_t)wv * 16 * HD, HD, nl, hf, 32);
  SFENCE();
  for (int st = wv; st < 64; st += 4) {
    int stn = (st + 4 < 64) ? st + 4 : st;   // branch-free prefetch index
    v16bf nk0 = load_frag_b(Kb + (size_t)stn * 16 * HD, HD, nl, hf, 0);
    v16bf nk1 = load_frag_b(Kb + (size_t)stn * 16 * HD, HD, nl, hf, 32);
    v8f acc = {};
    acc = wmma_bf16(aq0, bk0, acc);
    acc = wmma_bf16(aq1, bk1, acc);
#pragma unroll
    for (int r = 0; r < 8; ++r)
      sc[(hf * 8 + r) * 1024 + st * 16 + nl] = acc[r];
    bk0 = nk0; bk1 = nk1;
    // pin interleave: prefetch loads between the 2 WMMAs, stores last
    SGB(0x020, 2); SGB(0x008, 1);
    SGB(0x020, 2); SGB(0x008, 1);
    SGB(0x200, 8);
  }
  __syncthreads();

  // ---- Stage 2: row softmax (16 rows x 1024, 8 threads per row) ----
  int i = threadIdx.x >> 3, seg = threadIdx.x & 7;
  float* rowp = sc + i * 1024 + seg * 128;
  float m = -3.0e38f;
#pragma unroll 8
  for (int j = 0; j < 128; ++j) m = fmaxf(m, rowp[j]);
  red[i * 8 + seg] = m;
  __syncthreads();
  float M = red[i * 8];
#pragma unroll
  for (int j = 1; j < 8; ++j) M = fmaxf(M, red[i * 8 + j]);
  __syncthreads();
  float s = 0.0f;
#pragma unroll 8
  for (int j = 0; j < 128; ++j) {
    float p = __expf(rowp[j] - M);
    rowp[j] = p;
    s += p;
  }
  red[i * 8 + seg] = s;
  __syncthreads();
  float S = 0.0f;
#pragma unroll
  for (int j = 0; j < 8; ++j) S += red[i * 8 + j];
  float inv = 1.0f / S;
  bf16_t* prow = P + i * 1024 + seg * 128;
#pragma unroll 8
  for (int j = 0; j < 128; ++j) prow[j] = (bf16_t)(rowp[j] * inv);
  __syncthreads();

  // ---- Stage 3: O = P V ; wave wv owns 16 output columns ----
  // A from LDS (DScnt), B from global (LOADcnt): independent counters,
  // software-pipelined one k-step ahead with pinned interleave.
  const bf16_t* Vcol = Vb + (size_t)(wv * 16) * LL;   // Vt rows = head dims
  v8f o = {};
  v16bf pa = load_frag_a(P, 1024, nl, hf, 0);
  v16bf vb = load_frag_b(Vcol, LL, nl, hf, 0);
  SFENCE();
#pragma unroll 4
  for (int k0 = 32; k0 < LL; k0 += 32) {
    v16bf pan = load_frag_a(P, 1024, nl, hf, k0);     // ds_load_b128
    v16bf vbn = load_frag_b(Vcol, LL, nl, hf, k0);    // global_load_b128
    o = wmma_bf16(pa, vb, o);
    pa = pan; vb = vbn;
    SGB(0x100, 2); SGB(0x020, 2); SGB(0x008, 1);
  }
  o = wmma_bf16(pa, vb, o);
#pragma unroll
  for (int r = 0; r < 8; ++r) {
    int l = qt * 16 + hf * 8 + r;
    AO[((size_t)l * NN + b_) * EE + h * HD + wv * 16 + nl] = (bf16_t)o[r];
  }
}

// ---------------------------------------------------------------------------
// Kernel 4: out projection (8192 x 512) = AO (8192x512) * Wout^T + bias -> f32
// ---------------------------------------------------------------------------
__global__ __launch_bounds__(256) void out_proj_kernel(
    const bf16_t* __restrict__ AO, const bf16_t* __restrict__ WoutB,
    const float* __restrict__ bias, float* __restrict__ out) {
  int wave = blockIdx.x * 8 + (threadIdx.x >> 5);
  int lane = threadIdx.x & 31;
  int nl = lane & 15, hf = lane >> 4;
  int mtile = wave / 8;               // 512 m-tiles
  int strip = wave % 8;               // 8 strips of 64 cols
  int m0 = mtile * 16, c0 = strip * 64;

  const bf16_t* Ab  = AO + (size_t)m0 * EE;
  const bf16_t* Bb0 = WoutB + (size_t)(c0 +  0) * EE;
  const bf16_t* Bb1 = WoutB + (size_t)(c0 + 16) * EE;
  const bf16_t* Bb2 = WoutB + (size_t)(c0 + 32) * EE;
  const bf16_t* Bb3 = WoutB + (size_t)(c0 + 48) * EE;

  v8f acc0 = {}, acc1 = {}, acc2 = {}, acc3 = {};
  v16bf a  = load_frag_a(Ab, EE, nl, hf, 0);
  v16bf b0 = load_frag_b(Bb0, EE, nl, hf, 0);
  v16bf b1 = load_frag_b(Bb1, EE, nl, hf, 0);
  v16bf b2 = load_frag_b(Bb2, EE, nl, hf, 0);
  v16bf b3 = load_frag_b(Bb3, EE, nl, hf, 0);
  SFENCE();

#pragma unroll
  for (int k0 = 32; k0 < EE; k0 += 32) {
    v16bf an  = load_frag_a(Ab, EE, nl, hf, k0);
    v16bf bn0 = load_frag_b(Bb0, EE, nl, hf, k0);
    v16bf bn1 = load_frag_b(Bb1, EE, nl, hf, k0);
    v16bf bn2 = load_frag_b(Bb2, EE, nl, hf, k0);
    v16bf bn3 = load_frag_b(Bb3, EE, nl, hf, k0);
    acc0 = wmma_bf16(a, b0, acc0);
    acc1 = wmma_bf16(a, b1, acc1);
    acc2 = wmma_bf16(a, b2, acc2);
    acc3 = wmma_bf16(a, b3, acc3);
    a = an; b0 = bn0; b1 = bn1; b2 = bn2; b3 = bn3;
    SGB(0x020, 2); SGB(0x008, 1);
    SGB(0x020, 2); SGB(0x008, 1);
    SGB(0x020, 2); SGB(0x008, 1);
    SGB(0x020, 2); SGB(0x008, 1);
    SGB(0x020, 2);
  }
  acc0 = wmma_bf16(a, b0, acc0);
  acc1 = wmma_bf16(a, b1, acc1);
  acc2 = wmma_bf16(a, b2, acc2);
  acc3 = wmma_bf16(a, b3, acc3);

  v8f accs[4] = {acc0, acc1, acc2, acc3};
#pragma unroll
  for (int t = 0; t < 4; ++t) {
    int c = c0 + t * 16 + nl;
    float bv = bias[c];
#pragma unroll
    for (int r = 0; r < 8; ++r) {
      int row = m0 + hf * 8 + r;
      out[(size_t)row * EE + c] = accs[t][r] + bv;
    }
  }
}

// ---------------------------------------------------------------------------
// Host launcher
// ---------------------------------------------------------------------------
extern "C" void kernel_launch(void* const* d_in, const int* in_sizes, int n_in,
                              void* d_out, int out_size, void* d_ws,
                              size_t ws_size, hipStream_t stream) {
  const float* query = (const float*)d_in[0];
  const float* timev = (const float*)d_in[1];
  const float* W_in  = (const float*)d_in[2];
  const float* b_in  = (const float*)d_in[3];
  const float* W_out = (const float*)d_in[4];
  const float* b_out = (const float*)d_in[5];
  float* out = (float*)d_out;

  char* ws = (char*)d_ws;
  size_t off = 0;
  bf16_t* XB    = (bf16_t*)(ws + off); off += (size_t)N_XB   * 2;  // 8 MB
  bf16_t* WinB  = (bf16_t*)(ws + off); off += (size_t)N_WIN  * 2;  // 1.5 MB
  bf16_t* WoutB = (bf16_t*)(ws + off); off += (size_t)N_WOUT * 2;  // 0.5 MB
  bf16_t* Q     = (bf16_t*)(ws + off); off += (size_t)64 * LL * HD * 2; // 8 MB
  bf16_t* K     = (bf16_t*)(ws + off); off += (size_t)64 * LL * HD * 2; // 8 MB
  bf16_t* Vt    = (bf16_t*)(ws + off); off += (size_t)64 * LL * HD * 2; // 8 MB
  bf16_t* AO    = (bf16_t*)(ws + off); off += (size_t)RR * EE * 2;      // 8 MB
  float*  ep    = (float*)(ws + off);  off += (size_t)RR * 4;
  float*  en    = (float*)(ws + off);  off += (size_t)RR * 4;

  // 1) precision convert + temporal factors
  convert_kernel<<<(N_CVT + 255) / 256, 256, 0, stream>>>(
      query, W_in, W_out, timev, XB, WinB, WoutB, ep, en);

  // 2) QKV projection: 512 mtiles * 24 strips = 12288 waves -> 1536 blocks
  qkv_gemm_kernel<<<1536, 256, 0, stream>>>(XB, WinB, b_in, ep, en, Q, K, Vt);

  // 3) attention: 64 (b,h) * 64 q-tiles = 4096 blocks, 128 threads, ~97KB LDS
  attn_kernel<<<4096, 128, SMEM_ATTN, stream>>>(Q, K, Vt, AO);

  // 4) out projection: 512 mtiles * 8 strips = 4096 waves -> 512 blocks
  out_proj_kernel<<<512, 256, 0, stream>>>(AO, WoutB, b_out, out);
}
